// ContextType_76768245449670
// MI455X (gfx1250) — matrix operand order repaired and checked by reading, not compiled, not run on hardware
//
#include <hip/hip_runtime.h>
#include <hip/hip_bf16.h>
#include <math.h>

// Problem constants (match reference)
#define TT   8192      // sequence length
#define EE   256       // embedding dim
#define HH   256       // hidden dim
#define G4   1024      // 4*H gate rows
#define NWG  8         // workgroups in the scan phase
#define ROWS 128       // gate rows per scan workgroup
#define WSTR 257       // padded LDS row stride (bank-conflict free)

typedef __attribute__((ext_vector_type(16))) __bf16 v16bf;
typedef __attribute__((ext_vector_type(8)))  __bf16 v8bf;
typedef __attribute__((ext_vector_type(8)))  float  v8f;

// ---------------------------------------------------------------------------
// Prep kernel A: gather embedding rows once and split fp32 -> bf16 hi/lo.
// One block per token row; 256 threads = 256 elements.
// ---------------------------------------------------------------------------
__global__ __launch_bounds__(256)
void prep_emb(const int*   __restrict__ tokens,
              const float* __restrict__ emb,   // [V, 256]
              __bf16*      __restrict__ Ahi,   // [T, 256]
              __bf16*      __restrict__ Alo)   // [T, 256]
{
    const int row = blockIdx.x;                // 0..T-1
    const int k   = threadIdx.x;               // 0..255
    const size_t tok = (size_t)tokens[row];
    const float  v   = emb[tok * EE + k];
    const __bf16 hi  = (__bf16)v;
    Ahi[(size_t)row * EE + k] = hi;
    Alo[(size_t)row * EE + k] = (__bf16)(v - (float)hi);
}

// ---------------------------------------------------------------------------
// Prep kernel B: split W_ih fp32 -> bf16 hi/lo (done once, L2-resident after).
// ---------------------------------------------------------------------------
__global__ __launch_bounds__(256)
void prep_wih(const float* __restrict__ Wih,   // [1024, 256]
              __bf16*      __restrict__ Whi,
              __bf16*      __restrict__ Wlo)
{
    const size_t i = (size_t)blockIdx.x * 256 + threadIdx.x;  // < 1024*256
    const float  v  = Wih[i];
    const __bf16 hi = (__bf16)v;
    Whi[i] = hi;
    Wlo[i] = (__bf16)(v - (float)hi);
}

// ---------------------------------------------------------------------------
// Kernel 1: xg = emb[tokens] @ W_ih^T + (b_ih + b_hh), compensated bf16 WMMA.
// Operands are pre-split bf16, so each fragment is two global_load_b128s.
// Grid: (2, 512) blocks of 256 threads (8 waves). Wave w handles 4 col-tiles.
// ---------------------------------------------------------------------------
__global__ __launch_bounds__(256)
void xg_gemm_wmma(const __bf16* __restrict__ Ahi,  // [T, 256]
                  const __bf16* __restrict__ Alo,
                  const __bf16* __restrict__ Whi,  // [1024, 256]
                  const __bf16* __restrict__ Wlo,
                  const float*  __restrict__ bih,
                  const float*  __restrict__ bhh,
                  float*        __restrict__ xg)   // [T, 1024]
{
    const int wave = threadIdx.x >> 5;
    const int lane = threadIdx.x & 31;
    const int half = lane >> 4;       // lane half (K-split for A/B fragments)
    const int l16  = lane & 15;       // M (A rows) / N (B cols / C cols)

    const int tm  = blockIdx.y;                      // row tile 0..511
    const int tn0 = (blockIdx.x * 8 + wave) * 4;     // 4 col tiles per wave

    const int row = tm * 16 + l16;
    const __bf16* arow_hi = Ahi + (size_t)row * EE;
    const __bf16* arow_lo = Alo + (size_t)row * EE;

    v8f acc[4] = {};   // 4 col-tiles of 16x16 f32 accumulators

    for (int kb = 0; kb < 8; ++kb) {
        const int k0 = kb * 32;

        // A fragment 16x32 bf16 (ISA 7.12.2): per lane, elements [0..7] are
        // K = k0 + 8*half + (0..7), elements [8..15] are K = k0+16+8*half+(0..7)
        // -> two contiguous 16B loads each for hi and lo.
        const int a0 = k0 + 8 * half;
        const int a1 = k0 + 16 + 8 * half;
        const v8bf ah0 = *(const v8bf*)(arow_hi + a0);
        const v8bf ah1 = *(const v8bf*)(arow_hi + a1);
        const v8bf al0 = *(const v8bf*)(arow_lo + a0);
        const v8bf al1 = *(const v8bf*)(arow_lo + a1);
        const v16bf afhi = __builtin_shufflevector(ah0, ah1,
            0,1,2,3,4,5,6,7,8,9,10,11,12,13,14,15);
        const v16bf aflo = __builtin_shufflevector(al0, al1,
            0,1,2,3,4,5,6,7,8,9,10,11,12,13,14,15);

        #pragma unroll
        for (int j = 0; j < 4; ++j) {
            const int col = (tn0 + j) * 16 + l16;        // gate column (N)
            // B fragment 32x16 bf16: per lane, elements [0..7] are
            // K = k0 + 8*half + (0..7), [8..15] are K = k0+16+8*half+(0..7) of W_ih row `col` -> two 16B loads.
            const __bf16* bh = Whi + (size_t)col * EE + k0 + 8 * half;
            const __bf16* bl = Wlo + (size_t)col * EE + k0 + 8 * half;
            const v8bf bh0 = *(const v8bf*)bh;
            const v8bf bh1 = *(const v8bf*)(bh + 16);
            const v8bf bl0 = *(const v8bf*)bl;
            const v8bf bl1 = *(const v8bf*)(bl + 16);
            const v16bf bfhi = __builtin_shufflevector(bh0, bh1,
                0,1,2,3,4,5,6,7,8,9,10,11,12,13,14,15);
            const v16bf bflo = __builtin_shufflevector(bl0, bl1,
                0,1,2,3,4,5,6,7,8,9,10,11,12,13,14,15);

            // D = Alo*Bhi + Ahi*Blo + Ahi*Bhi + C   (compensated bf16)
            acc[j] = __builtin_amdgcn_wmma_f32_16x16x32_bf16(
                         false, aflo, false, bfhi, (short)0, acc[j], false, false);
            acc[j] = __builtin_amdgcn_wmma_f32_16x16x32_bf16(
                         false, afhi, false, bflo, (short)0, acc[j], false, false);
            acc[j] = __builtin_amdgcn_wmma_f32_16x16x32_bf16(
                         false, afhi, false, bfhi, (short)0, acc[j], false, false);
        }
    }

    // Epilogue: C/D layout — lane<16: N=l16, M=j; lane>=16: M=8+j
    #pragma unroll
    for (int j = 0; j < 4; ++j) {
        const int col = (tn0 + j) * 16 + l16;
        const float bias = bih[col] + bhh[col];
        #pragma unroll
        for (int m = 0; m < 8; ++m) {
            const int r = tm * 16 + m + 8 * half;
            xg[(size_t)r * G4 + col] = acc[j][m] + bias;
        }
    }
}

// ---------------------------------------------------------------------------
// Kernel 2: persistent 8-WG LSTM scan. WG w owns gate rows [128w,128w+128)
// with its W_hh rows resident in LDS. One global spin-barrier per step.
// Every WG keeps a redundant copy of h (LDS) and c (one element per thread).
// ---------------------------------------------------------------------------
__global__ __launch_bounds__(256)
void lstm_scan(const float* __restrict__ Whh,      // [1024, 256]
               const float* __restrict__ xg,       // [T, 1024]
               float*       __restrict__ out,      // [T*256 + 256 + 256]
               float*       __restrict__ gatebuf,  // [2][1024] (global scratch)
               unsigned int* __restrict__ sync_cnt)
{
    extern __shared__ float smem[];
    float* WT      = smem;                     // [ROWS][WSTR] row-major W_hh slice (padded rows)
    float* hbuf    = smem + ROWS * WSTR;       // [256]
    float* partial = hbuf + HH;                // [256]

    const int wg   = blockIdx.x;               // 0..7
    const int tid  = threadIdx.x;              // 0..255
    const int gate = wg >> 1;                  // 0:i 1:f 2:g 3:o

    // Stage this WG's 128 weight rows into LDS: WT[o][k] = Whh[128*wg + o][k]
    for (int i = tid; i < ROWS * HH; i += 256) {
        const int o = i >> 8;                  // 0..127
        const int k = i & 255;                 // 0..255
        WT[o * WSTR + k] = Whh[(size_t)(ROWS * wg + o) * HH + k];
    }
    hbuf[tid] = 0.0f;
    float c = 0.0f;                            // this thread owns c[tid]
    __syncthreads();

    const int o     = tid & 127;               // output row within WG
    const int half  = tid >> 7;                // K-half: 0 -> [0,128), 1 -> [128,256)
    const int kbase = half * 128;

    for (int t = 0; t < TT; ++t) {
        // Load this step's xg slice; prefetch next step's slice into L2/WGP$.
        float xgv = 0.0f;
        if (tid < 128) {
            xgv = xg[(size_t)t * G4 + ROWS * wg + tid];
            if (t + 1 < TT)
                __builtin_prefetch(xg + (size_t)(t + 1) * G4 + ROWS * wg + tid, 0, 0);
        }

        // ---- half-matvec: 128 FMAs from LDS (conflict-free, h broadcast) ----
        float sum = 0.0f;
        #pragma unroll 8
        for (int k = 0; k < 128; ++k)
            sum = fmaf(WT[o * WSTR + kbase + k], hbuf[kbase + k], sum);
        partial[tid] = sum;
        __syncthreads();

        // ---- combine halves, apply gate nonlinearity, publish to L2 ----
        if (tid < 128) {
            const float raw = partial[tid] + partial[tid + 128] + xgv;
            const float act = (gate == 2) ? tanhf(raw)
                                          : 1.0f / (1.0f + expf(-raw));
            __hip_atomic_store(&gatebuf[(size_t)(t & 1) * G4 + ROWS * wg + tid],
                               act, __ATOMIC_RELAXED, __HIP_MEMORY_SCOPE_AGENT);
        }
        __threadfence();        // make this thread's stores device-visible
        __syncthreads();

        // ---- global barrier: release-arrive, acquire-spin (monotonic) ----
        if (tid == 0) {
            __hip_atomic_fetch_add(sync_cnt, 1u, __ATOMIC_ACQ_REL,
                                   __HIP_MEMORY_SCOPE_AGENT);
            const unsigned target = (unsigned)(NWG * (t + 1));
            while (__hip_atomic_load(sync_cnt, __ATOMIC_ACQUIRE,
                                     __HIP_MEMORY_SCOPE_AGENT) < target)
                __builtin_amdgcn_s_sleep(1);
        }
        __syncthreads();

        // ---- elementwise cell update (redundant in every WG) ----
        const float* gb = gatebuf + (size_t)(t & 1) * G4;
        const float iv = __hip_atomic_load(&gb[tid],       __ATOMIC_RELAXED, __HIP_MEMORY_SCOPE_AGENT);
        const float fv = __hip_atomic_load(&gb[256 + tid], __ATOMIC_RELAXED, __HIP_MEMORY_SCOPE_AGENT);
        const float gv = __hip_atomic_load(&gb[512 + tid], __ATOMIC_RELAXED, __HIP_MEMORY_SCOPE_AGENT);
        const float ov = __hip_atomic_load(&gb[768 + tid], __ATOMIC_RELAXED, __HIP_MEMORY_SCOPE_AGENT);
        c = fv * c + iv * gv;
        const float hn = ov * tanhf(c);
        hbuf[tid] = hn;
        if (wg == 0) out[(size_t)t * HH + tid] = hn;   // hiddens[t]
        __syncthreads();                                // h ready for next matvec
    }

    if (wg == 0) {
        out[(size_t)TT * HH + tid]      = hbuf[tid];   // h_last
        out[(size_t)TT * HH + HH + tid] = c;           // c_last
    }
}

// ---------------------------------------------------------------------------
extern "C" void kernel_launch(void* const* d_in, const int* in_sizes, int n_in,
                              void* d_out, int out_size, void* d_ws, size_t ws_size,
                              hipStream_t stream) {
    const int*   tokens = (const int*)  d_in[0];
    const float* emb    = (const float*)d_in[1];
    const float* Wih    = (const float*)d_in[2];
    const float* Whh    = (const float*)d_in[3];
    const float* bih    = (const float*)d_in[4];
    const float* bhh    = (const float*)d_in[5];
    float* out = (float*)d_out;

    // Workspace layout (all 16B-aligned):
    //   [0, 256)                     : global barrier counter
    //   [256, 256+8KB)               : gate ping-pong buffer [2][1024] f32
    //   [16KB, 16KB+32MB)            : xg [8192][1024] f32
    //   ... +4MB                     : Ahi bf16 [8192][256]
    //   ... +4MB                     : Alo bf16 [8192][256]
    //   ... +512KB                   : Whi bf16 [1024][256]
    //   ... +512KB                   : Wlo bf16 [1024][256]
    unsigned char* ws = (unsigned char*)d_ws;
    unsigned int* sync_cnt = (unsigned int*)ws;
    float*  gatebuf = (float*)(ws + 256);
    float*  xg      = (float*)(ws + 16384);
    size_t  off     = 16384 + (size_t)TT * G4 * sizeof(float);
    __bf16* Ahi = (__bf16*)(ws + off);  off += (size_t)TT * EE * sizeof(__bf16);
    __bf16* Alo = (__bf16*)(ws + off);  off += (size_t)TT * EE * sizeof(__bf16);
    __bf16* Whi = (__bf16*)(ws + off);  off += (size_t)G4 * EE * sizeof(__bf16);
    __bf16* Wlo = (__bf16*)(ws + off);

    hipMemsetAsync(ws, 0, 256, stream);   // reset barrier counter each launch

    // Phase 0: one-time operand preparation (gather + bf16 hi/lo split)
    prep_emb<<<dim3(TT), 256, 0, stream>>>(tokens, emb, Ahi, Alo);
    prep_wih<<<dim3(G4), 256, 0, stream>>>(Wih, Whi, Wlo);

    // Phase 1: WMMA GEMM for all input-gate contributions
    dim3 ggrid(2, TT / 16, 1);
    xg_gemm_wmma<<<ggrid, 256, 0, stream>>>(Ahi, Alo, Whi, Wlo, bih, bhh, xg);

    // Phase 2: persistent 8-WG recurrence (dynamic LDS: 128*257 + 2*256 floats)
    const size_t shmem = (size_t)(ROWS * WSTR + 2 * HH) * sizeof(float);
    lstm_scan<<<dim3(NWG), 256, shmem, stream>>>(Whh, xg, out, gatebuf, sync_cnt);
}
